// MultiHeadSelfAttention_59528246722880
// MI455X (gfx1250) — matrix operand order repaired and checked by reading, not compiled, not run on hardware
//
#include <hip/hip_runtime.h>
#include <hip/hip_bf16.h>
#include <stdint.h>

// ---------- types ----------
typedef __attribute__((ext_vector_type(16))) __bf16        v16bf;
typedef __attribute__((ext_vector_type(8)))  float         v8f;
typedef __attribute__((ext_vector_type(4)))  float         f32x4;
typedef __attribute__((ext_vector_type(4)))  unsigned int  u32x4;

union FragBF {
    v16bf v;
    u32x4 q[2];
};

__device__ __forceinline__ unsigned short f32_to_bf16(float f) {
    unsigned int u = __float_as_uint(f);
    unsigned int r = 0x7FFFu + ((u >> 16) & 1u);   // round-to-nearest-even
    return (unsigned short)((u + r) >> 16);
}

__device__ __forceinline__ unsigned int pack_bf16x2(float lo, float hi) {
    return (unsigned int)f32_to_bf16(lo) | ((unsigned int)f32_to_bf16(hi) << 16);
}

__device__ __forceinline__ v8f v8f_zero() {
    v8f z = {0.f, 0.f, 0.f, 0.f, 0.f, 0.f, 0.f, 0.f};
    return z;
}

__device__ __forceinline__ v8f wmma_bf16(const FragBF& a, const FragBF& b, v8f c) {
    // (neg_a, A, neg_b, B, c_mod, C, reuse_a, reuse_b)
    return __builtin_amdgcn_wmma_f32_16x16x32_bf16(false, a.v, false, b.v,
                                                   (short)0, c, false, false);
}

// Problem constants
#define BB 4
#define SS 2048
#define DD 1024
#define HH 16
#define DH 64

// ---------- fp32 -> bf16 conversion (8 elements / thread) ----------
__global__ void k_cvt_bf16(const float* __restrict__ src,
                           unsigned short* __restrict__ dst, int n) {
    int i = (blockIdx.x * blockDim.x + threadIdx.x) * 8;
    if (i < n) {
        f32x4 x0 = *(const f32x4*)(src + i);
        f32x4 x1 = *(const f32x4*)(src + i + 4);
        u32x4 o;
        o.x = pack_bf16x2(x0.x, x0.y);
        o.y = pack_bf16x2(x0.z, x0.w);
        o.z = pack_bf16x2(x1.x, x1.y);
        o.w = pack_bf16x2(x1.z, x1.w);
        *(u32x4*)(dst + i) = o;
    }
}

// ---------- GEMM: C[M=8192, N=1024] = A(bf16 MxK) @ W(bf16 NxK)^T + bias ----------
// Wave tile: 32(M) x 64(N): each B fragment feeds two wmmas; k-loop is
// ping-pong software-pipelined (loads for k+32 issued before wmmas of k).
// mode 0: bf16 out, [B,H,S,Dh] scatter    (Q, K)
// mode 1: bf16 out, [B,H,Dh,S] scatter    (V transposed)
// mode 2: f32  out, row-major [M,N]       (final projection -> d_out)
__global__ void __launch_bounds__(128)
k_gemm_nt(const unsigned short* __restrict__ A,
          const unsigned short* __restrict__ W,
          const float* __restrict__ bias,
          void* __restrict__ dst, int mode) {
    const int K = DD;
    const int wave = threadIdx.x >> 5;
    const int lane = threadIdx.x & 31;
    const int l16  = lane & 15;
    const int lh   = lane >> 4;

    const int wid = blockIdx.x * 4 + wave;   // 4096 wave-tiles
    const int m0  = (wid >> 4) << 5;         // 256 m-tiles of 32
    const int n0  = (wid & 15) << 6;         // 16  n-strips of 64

    v8f acc[2][4];
#pragma unroll
    for (int r = 0; r < 2; ++r)
#pragma unroll
        for (int t = 0; t < 4; ++t) acc[r][t] = v8f_zero();

    const unsigned short* arow[2];
#pragma unroll
    for (int r = 0; r < 2; ++r)
        arow[r] = A + (size_t)(m0 + r * 16 + l16) * K + lh * 16;
    const unsigned short* wrow[4];
#pragma unroll
    for (int t = 0; t < 4; ++t)
        wrow[t] = W + (size_t)(n0 + t * 16 + l16) * K + lh * 16;

    FragBF aP[2], aQ[2], bP[4], bQ[4];

    // prologue: stage k = 0
#pragma unroll
    for (int r = 0; r < 2; ++r) {
        aP[r].q[0] = *(const u32x4*)(arow[r]);
        aP[r].q[1] = *(const u32x4*)(arow[r] + 8);
    }
#pragma unroll
    for (int t = 0; t < 4; ++t) {
        bP[t].q[0] = *(const u32x4*)(wrow[t]);
        bP[t].q[1] = *(const u32x4*)(wrow[t] + 8);
    }

    for (int kk = 0; kk < K - 64; kk += 64) {
        // prefetch stage kk+32
#pragma unroll
        for (int r = 0; r < 2; ++r) {
            aQ[r].q[0] = *(const u32x4*)(arow[r] + kk + 32);
            aQ[r].q[1] = *(const u32x4*)(arow[r] + kk + 40);
        }
#pragma unroll
        for (int t = 0; t < 4; ++t) {
            bQ[t].q[0] = *(const u32x4*)(wrow[t] + kk + 32);
            bQ[t].q[1] = *(const u32x4*)(wrow[t] + kk + 40);
        }
        // compute stage kk
#pragma unroll
        for (int t = 0; t < 4; ++t) {
            acc[0][t] = wmma_bf16(aP[0], bP[t], acc[0][t]);
            acc[1][t] = wmma_bf16(aP[1], bP[t], acc[1][t]);
        }
        // prefetch stage kk+64
#pragma unroll
        for (int r = 0; r < 2; ++r) {
            aP[r].q[0] = *(const u32x4*)(arow[r] + kk + 64);
            aP[r].q[1] = *(const u32x4*)(arow[r] + kk + 72);
        }
#pragma unroll
        for (int t = 0; t < 4; ++t) {
            bP[t].q[0] = *(const u32x4*)(wrow[t] + kk + 64);
            bP[t].q[1] = *(const u32x4*)(wrow[t] + kk + 72);
        }
        // compute stage kk+32
#pragma unroll
        for (int t = 0; t < 4; ++t) {
            acc[0][t] = wmma_bf16(aQ[0], bQ[t], acc[0][t]);
            acc[1][t] = wmma_bf16(aQ[1], bQ[t], acc[1][t]);
        }
    }
    // tail: stages K-64 (in P) and K-32
    {
        const int kk = K - 64;
#pragma unroll
        for (int r = 0; r < 2; ++r) {
            aQ[r].q[0] = *(const u32x4*)(arow[r] + kk + 32);
            aQ[r].q[1] = *(const u32x4*)(arow[r] + kk + 40);
        }
#pragma unroll
        for (int t = 0; t < 4; ++t) {
            bQ[t].q[0] = *(const u32x4*)(wrow[t] + kk + 32);
            bQ[t].q[1] = *(const u32x4*)(wrow[t] + kk + 40);
        }
#pragma unroll
        for (int t = 0; t < 4; ++t) {
            acc[0][t] = wmma_bf16(aP[0], bP[t], acc[0][t]);
            acc[1][t] = wmma_bf16(aP[1], bP[t], acc[1][t]);
        }
#pragma unroll
        for (int t = 0; t < 4; ++t) {
            acc[0][t] = wmma_bf16(aQ[0], bQ[t], acc[0][t]);
            acc[1][t] = wmma_bf16(aQ[1], bQ[t], acc[1][t]);
        }
    }

    // epilogue
#pragma unroll
    for (int t = 0; t < 4; ++t) {
        const int j  = n0 + t * 16 + l16;
        const float bj = bias[j];
#pragma unroll
        for (int r = 0; r < 2; ++r) {
#pragma unroll
            for (int v = 0; v < 8; ++v) {
                const int i = m0 + r * 16 + v + lh * 8;
                const float val = acc[r][t][v] + bj;
                if (mode == 2) {
                    ((float*)dst)[(size_t)i * DD + j] = val;
                } else {
                    const int b_ = i >> 11, s_ = i & (SS - 1);
                    const int h_ = j >> 6,  d_ = j & (DH - 1);
                    const unsigned short ov = f32_to_bf16(val);
                    if (mode == 0)
                        ((unsigned short*)dst)[(((size_t)b_ * HH + h_) * SS + s_) * DH + d_] = ov;
                    else
                        ((unsigned short*)dst)[(((size_t)b_ * HH + h_) * DH + d_) * SS + s_] = ov;
                }
            }
        }
    }
}

// ---------- flash attention ----------
// grid: B*H*(S/16) blocks, 128 threads (4 waves). Each wave owns key chunks of 32
// (interleaved stride 128). Q,K in [B,H,S,Dh] bf16; V in [B,H,Dh,S] bf16.
// Output: bf16 concat [B,S,D].
__global__ void __launch_bounds__(128)
k_attn(const unsigned short* __restrict__ Qm,
       const unsigned short* __restrict__ Km,
       const unsigned short* __restrict__ Vt,
       const unsigned char* __restrict__ pad,
       unsigned short* __restrict__ Cout) {
    __shared__ float        s_stage[4][16][32];   // raw masked/scaled scores
    __shared__ unsigned int s_p[4][16][16];       // bf16 probabilities (packed)
    __shared__ float        s_alpha[4][16];
    __shared__ float        s_m[4][16], s_l[4][16], s_beta[4][16];
    __shared__ float        s_L[16];
    __shared__ float        s_Ow[4][16][DH];      // per-wave partial O

    const int tid  = threadIdx.x;
    const int wave = tid >> 5, lane = tid & 31;
    const int l16  = lane & 15, lh = lane >> 4;

    const int qt = blockIdx.x & 127;
    const int h  = (blockIdx.x >> 7) & 15;
    const int b  = blockIdx.x >> 11;
    const int q0 = qt << 4;
    const size_t bh = (size_t)b * HH + h;

    const unsigned char* pmb = pad + (size_t)b * SS;
    unsigned int qmask = 0;
#pragma unroll
    for (int r = 0; r < 16; ++r) qmask |= (pmb[q0 + r] ? 1u : 0u) << r;

    // preload Q fragments (A operand) for both k-steps over Dh
    const unsigned short* qrow = Qm + (bh * SS + q0 + l16) * DH;
    FragBF aq[2];
#pragma unroll
    for (int ks = 0; ks < 2; ++ks) {
        aq[ks].q[0] = *(const u32x4*)(qrow + ks * 32 + lh * 16);
        aq[ks].q[1] = *(const u32x4*)(qrow + ks * 32 + lh * 16 + 8);
    }

    v8f o[4];
#pragma unroll
    for (int t = 0; t < 4; ++t) o[t] = v8f_zero();
    float m_run = -1e30f, l_run = 0.f;

    for (int ck = wave * 32; ck < SS; ck += 128) {
        // ---- load all K fragments for this chunk, then run 4 score wmmas ----
        FragBF kb[4];
#pragma unroll
        for (int t = 0; t < 2; ++t) {
            const unsigned short* krow = Km + (bh * SS + ck + t * 16 + l16) * DH;
#pragma unroll
            for (int ks = 0; ks < 2; ++ks) {
                kb[t * 2 + ks].q[0] = *(const u32x4*)(krow + ks * 32 + lh * 16);
                kb[t * 2 + ks].q[1] = *(const u32x4*)(krow + ks * 32 + lh * 16 + 8);
            }
        }
        v8f cs[2];
        cs[0] = v8f_zero(); cs[1] = v8f_zero();
        cs[0] = wmma_bf16(aq[0], kb[0], cs[0]);
        cs[0] = wmma_bf16(aq[1], kb[1], cs[0]);
        cs[1] = wmma_bf16(aq[0], kb[2], cs[1]);
        cs[1] = wmma_bf16(aq[1], kb[3], cs[1]);

        // ---- prefetch V fragments now; softmax VALU below hides their latency ----
        FragBF vb[4];
#pragma unroll
        for (int t = 0; t < 4; ++t) {
            const unsigned short* vrow =
                Vt + (bh * DH + t * 16 + l16) * (size_t)SS + ck + lh * 16;
            vb[t].q[0] = *(const u32x4*)(vrow);
            vb[t].q[1] = *(const u32x4*)(vrow + 8);
        }

        // ---- scale (1/Dh), pad mask, stage (also transposes C->A layout) ----
#pragma unroll
        for (int t = 0; t < 2; ++t) {
            const int key = ck + t * 16 + l16;
            const bool km = pmb[key] != 0;
#pragma unroll
            for (int v = 0; v < 8; ++v) {
                const int row = v + lh * 8;
                float s = cs[t][v] * (1.0f / (float)DH);
                if (km || ((qmask >> row) & 1u)) s = -1e8f;
                s_stage[wave][row][t * 16 + l16] = s;
            }
        }
        __builtin_amdgcn_wave_barrier();
        // ---- online softmax (lanes 0..15, one row each) ----
        if (lh == 0) {
            const int r = l16;
            float tmax = -1e30f;
#pragma unroll
            for (int i = 0; i < 32; ++i) tmax = fmaxf(tmax, s_stage[wave][r][i]);
            const float m_new = fmaxf(m_run, tmax);
            const float alpha = __expf(m_run - m_new);
            float sum = 0.f;
#pragma unroll
            for (int i = 0; i < 16; ++i) {
                const float p0 = __expf(s_stage[wave][r][2 * i]     - m_new);
                const float p1 = __expf(s_stage[wave][r][2 * i + 1] - m_new);
                sum += p0 + p1;
                s_p[wave][r][i] = pack_bf16x2(p0, p1);
            }
            l_run = l_run * alpha + sum;
            m_run = m_new;
            s_alpha[wave][r] = alpha;
        }
        __builtin_amdgcn_wave_barrier();
        // ---- rescale O and accumulate P @ V ----
        float av[8];
#pragma unroll
        for (int v = 0; v < 8; ++v) av[v] = s_alpha[wave][v + lh * 8];
#pragma unroll
        for (int t = 0; t < 4; ++t)
#pragma unroll
            for (int v = 0; v < 8; ++v) o[t][v] *= av[v];

        FragBF pa;
        pa.q[0] = *(const u32x4*)&s_p[wave][l16][lh * 8];
        pa.q[1] = *(const u32x4*)&s_p[wave][l16][lh * 8 + 4];
#pragma unroll
        for (int t = 0; t < 4; ++t) o[t] = wmma_bf16(pa, vb[t], o[t]);
    }

    // ---- deterministic cross-wave combine ----
    if (lh == 0) { s_m[wave][l16] = m_run; s_l[wave][l16] = l_run; }
#pragma unroll
    for (int t = 0; t < 4; ++t)
#pragma unroll
        for (int v = 0; v < 8; ++v)
            s_Ow[wave][v + lh * 8][t * 16 + l16] = o[t][v];
    __syncthreads();
    if (tid < 16) {
        float M = s_m[0][tid];
        for (int w = 1; w < 4; ++w) M = fmaxf(M, s_m[w][tid]);
        float L = 0.f;
        for (int w = 0; w < 4; ++w) {
            const float bt = __expf(s_m[w][tid] - M);
            s_beta[w][tid] = bt;
            L += s_l[w][tid] * bt;
        }
        s_L[tid] = L;
    }
    __syncthreads();
    for (int i = tid; i < 16 * DH; i += 128) {
        const int row = i >> 6, d = i & (DH - 1);
        float val = 0.f;
        for (int w = 0; w < 4; ++w) val += s_Ow[w][row][d] * s_beta[w][row];
        val /= s_L[row];
        Cout[((size_t)b * SS + q0 + row) * DD + h * DH + d] = f32_to_bf16(val);
    }
}

// ---------- host launcher ----------
extern "C" void kernel_launch(void* const* d_in, const int* in_sizes, int n_in,
                              void* d_out, int out_size, void* d_ws, size_t ws_size,
                              hipStream_t stream) {
    (void)in_sizes; (void)n_in; (void)out_size; (void)ws_size;

    const float*         enc = (const float*)d_in[0];
    const unsigned char* pad = (const unsigned char*)d_in[1];  // numpy bool, 1B/elt
    const float* Wq = (const float*)d_in[2];
    const float* bq = (const float*)d_in[3];
    const float* Wk = (const float*)d_in[4];
    const float* bk = (const float*)d_in[5];
    const float* Wv = (const float*)d_in[6];
    const float* bv = (const float*)d_in[7];
    const float* Wo = (const float*)d_in[8];
    const float* bo = (const float*)d_in[9];

    char* ws = (char*)d_ws;
    const size_t MB = (size_t)1024 * 1024;
    unsigned short* Xbf  = (unsigned short*)(ws + 0 * MB);   // 16 MiB
    unsigned short* Wqbf = (unsigned short*)(ws + 16 * MB);  //  2 MiB
    unsigned short* Wkbf = (unsigned short*)(ws + 18 * MB);
    unsigned short* Wvbf = (unsigned short*)(ws + 20 * MB);
    unsigned short* Wobf = (unsigned short*)(ws + 22 * MB);
    unsigned short* Qbf  = (unsigned short*)(ws + 24 * MB);  // 16 MiB [B,H,S,Dh]
    unsigned short* Kbf  = (unsigned short*)(ws + 40 * MB);  // 16 MiB [B,H,S,Dh]
    unsigned short* Vtb  = (unsigned short*)(ws + 56 * MB);  // 16 MiB [B,H,Dh,S]
    unsigned short* Cbf  = (unsigned short*)(ws + 72 * MB);  // 16 MiB [B,S,D]

    const int nX = BB * SS * DD;   // 8388608
    const int nW = DD * DD;        // 1048576
    k_cvt_bf16<<<nX / 8 / 256, 256, 0, stream>>>(enc, Xbf, nX);
    k_cvt_bf16<<<nW / 8 / 256, 256, 0, stream>>>(Wq, Wqbf, nW);
    k_cvt_bf16<<<nW / 8 / 256, 256, 0, stream>>>(Wk, Wkbf, nW);
    k_cvt_bf16<<<nW / 8 / 256, 256, 0, stream>>>(Wv, Wvbf, nW);
    k_cvt_bf16<<<nW / 8 / 256, 256, 0, stream>>>(Wo, Wobf, nW);

    // Q = X Wq^T + bq ; K = X Wk^T + bk ; V = X Wv^T + bv (V stored transposed)
    k_gemm_nt<<<1024, 128, 0, stream>>>(Xbf, Wqbf, bq, (void*)Qbf, 0);
    k_gemm_nt<<<1024, 128, 0, stream>>>(Xbf, Wkbf, bk, (void*)Kbf, 0);
    k_gemm_nt<<<1024, 128, 0, stream>>>(Xbf, Wvbf, bv, (void*)Vtb, 1);

    // flash attention -> bf16 concat
    k_attn<<<BB * HH * (SS / 16), 128, 0, stream>>>(Qbf, Kbf, Vtb, pad, Cbf);

    // out = concat Wo^T + bo  (fp32)
    k_gemm_nt<<<1024, 128, 0, stream>>>(Cbf, Wobf, bo, d_out, 2);
}
